// GCNLayer_42949673129
// MI455X (gfx1250) — matrix operand order repaired and checked by reading, not compiled, not run on hardware
//
#include <hip/hip_runtime.h>
#include <hip/hip_bf16.h>

typedef __attribute__((ext_vector_type(16))) _Float16 v16h;
typedef __attribute__((ext_vector_type(8)))  _Float16 v8h;
typedef __attribute__((ext_vector_type(8)))  float    v8f;

#define DIM  128
#define KTOT 256   // 2*DIM

// ---------------------------------------------------------------------------
// Kernel 0: zero the agg accumulator (in d_out) and denom (in d_ws), and
// pre-convert W (f32 [128,256]) to f16 once per launch (L2-resident, 64KB).
// Removes all per-block f32->f16 converts from the WMMA hot loop.
// ---------------------------------------------------------------------------
__global__ void gat_prep_kernel(float* __restrict__ agg, float* __restrict__ denom,
                                const float* __restrict__ W, _Float16* __restrict__ Wh,
                                int nAgg, int nDen, int nW) {
    int i = blockIdx.x * blockDim.x + threadIdx.x;
    int stride = gridDim.x * blockDim.x;
    for (int j = i; j < nAgg; j += stride) agg[j] = 0.0f;
    for (int j = i; j < nDen; j += stride) denom[j] = 0.0f;
    for (int j = i; j < nW;   j += stride) Wh[j]   = (_Float16)W[j];
}

// ---------------------------------------------------------------------------
// Kernel 1: per-edge attention + scatter. One wave32 per edge.
// Lane l holds emb[.][4l..4l+3] (float4): coalesced 512B row reads, 5-step
// xor-shuffle reduction for the 128-dim dot, then f32 atomics into agg/denom.
// emb (25.6MB) and agg (25.6MB) are L2-resident (192MB L2) -> L2-atomic bound.
// ---------------------------------------------------------------------------
__global__ __launch_bounds__(256) void gat_edge_kernel(
    const float* __restrict__ emb, const int* __restrict__ row,
    const int* __restrict__ col, float* __restrict__ denom,
    float* __restrict__ agg, int E)
{
    int warpId = (blockIdx.x * blockDim.x + threadIdx.x) >> 5;
    int lane   = threadIdx.x & 31;
    if (warpId >= E) return;

    int r = row[warpId];
    int c = col[warpId];

    const float4 er = *(const float4*)(emb + (size_t)r * DIM + lane * 4);
    const float4 ec = *(const float4*)(emb + (size_t)c * DIM + lane * 4);

    float dot = er.x * ec.x + er.y * ec.y + er.z * ec.z + er.w * ec.w;
    #pragma unroll
    for (int off = 16; off > 0; off >>= 1)
        dot += __shfl_xor(dot, off, 32);          // wave32 reduction

    float attn = expf(dot);
    if (lane == 0) atomicAdd(denom + r, attn);

    float* ap = agg + (size_t)r * DIM + lane * 4;
    atomicAdd(ap + 0, attn * ec.x);
    atomicAdd(ap + 1, attn * ec.y);
    atomicAdd(ap + 2, attn * ec.z);
    atomicAdd(ap + 3, attn * ec.w);
}

// ---------------------------------------------------------------------------
// Kernel 2: per 16-node tile: normalize agg, build cat = [emb|agg] in LDS as
// f16, GEMM against W^T with v_wmma_f32_16x16x32_f16 (f32 accum), add bias,
// LayerNorm across the 128 outputs, write x to d_out (overwriting agg tile).
// Block = 128 threads = 4 waves; each wave owns 2 N-tiles of 16 outputs.
// B operands load straight from the pre-converted f16 W (2x b128 per tile).
// ---------------------------------------------------------------------------
__global__ __launch_bounds__(128) void gat_node_kernel(
    const float* __restrict__ emb, const _Float16* __restrict__ Wh,
    const float* __restrict__ bias, const float* __restrict__ gamma,
    const float* __restrict__ beta, const float* __restrict__ denom,
    float* __restrict__ out, int Ntot)
{
    __shared__ _Float16 catL[16][KTOT + 8];   // +8 halves row pad (bank spread)
    __shared__ float    xs[16][DIM + 4];      // +4 floats row pad
    __shared__ float    mu_s[16], rs_s[16];

    const int nodeBase = blockIdx.x * 16;
    const int tid = threadIdx.x;

    // Stage cat tile: 16 nodes x 256 features (emb || agg/denom), f32 -> f16.
    for (int idx = tid; idx < 16 * DIM; idx += 128) {
        int node = idx >> 7;
        int d    = idx & (DIM - 1);
        int n    = nodeBase + node; if (n >= Ntot) n = Ntot - 1;
        catL[node][d] = (_Float16)emb[(size_t)n * DIM + d];
        float dn = denom[n] + 1e-20f;
        catL[node][DIM + d] = (_Float16)(out[(size_t)n * DIM + d] / dn);
    }
    __syncthreads();

    const int wave  = tid >> 5;
    const int lane  = tid & 31;
    const int m     = lane & 15;
    const int hi    = lane >> 4;
    const int koffA = hi * 8;      // A operand: K runs {0..7,16..23} / {8..15,24..31}
    const int koffB = hi * 8;      // B operand: K runs {0..7,16..23} / {8..15,24..31}

    const int nt0  = wave * 2;                  // this wave's two N-tiles
    const int row0 = (nt0 + 0) * 16 + m;        // W rows == B columns
    const int row1 = (nt0 + 1) * 16 + m;
    const _Float16* wb0 = Wh + (size_t)row0 * KTOT + koffB;
    const _Float16* wb1 = Wh + (size_t)row1 * KTOT + koffB;
    __builtin_prefetch(wb0, 0, 1);              // global_prefetch_b8 (L2-resident W)
    __builtin_prefetch(wb1, 0, 1);

    v8f acc0 = {}; v8f acc1 = {};

    #pragma unroll
    for (int ks = 0; ks < 8; ++ks) {            // K = 256 in 8 steps of 32
        const int k0 = ks * 32;

        // A operand (cat tile, 16x32 f16) per ISA 16-bit A layout.
        v8h alo = *(const v8h*)&catL[m][k0 + koffA];
        v8h ahi = *(const v8h*)&catL[m][k0 + 16 + koffA];
        v16h a;
        #pragma unroll
        for (int i = 0; i < 8; ++i) { a[i] = alo[i]; a[8 + i] = ahi[i]; }

        // B operands: column n of B = row n of W; 8 halves at k0+koffB, 8 more at +16.
        v8h b0lo = *(const v8h*)(wb0 + k0);
        v8h b0hi = *(const v8h*)(wb0 + k0 + 16);
        v8h b1lo = *(const v8h*)(wb1 + k0);
        v8h b1hi = *(const v8h*)(wb1 + k0 + 16);
        v16h b0, b1;
        #pragma unroll
        for (int i = 0; i < 8; ++i) {
            b0[i] = b0lo[i]; b0[8 + i] = b0hi[i];
            b1[i] = b1lo[i]; b1[8 + i] = b1hi[i];
        }

        acc0 = __builtin_amdgcn_wmma_f32_16x16x32_f16(false, a, false, b0,
                                                      (short)0, acc0, false, false);
        acc1 = __builtin_amdgcn_wmma_f32_16x16x32_f16(false, a, false, b1,
                                                      (short)0, acc1, false, false);
    }

    // Spill D tiles (+bias) to LDS per C/D layout: VGPR v -> M = v + 8*hi, N = tile*16 + m.
    const float bv0 = bias[nt0 * 16 + m];
    const float bv1 = bias[(nt0 + 1) * 16 + m];
    #pragma unroll
    for (int v = 0; v < 8; ++v) {
        int Mr = v + 8 * hi;
        xs[Mr][nt0 * 16 + m]       = acc0[v] + bv0;
        xs[Mr][(nt0 + 1) * 16 + m] = acc1[v] + bv1;
    }
    __syncthreads();

    // LayerNorm stats: one leader thread per node row.
    if (tid < 16) {
        float s = 0.f, s2 = 0.f;
        #pragma unroll 4
        for (int d = 0; d < DIM; ++d) { float v = xs[tid][d]; s += v; s2 += v * v; }
        float mu  = s  * (1.0f / DIM);
        float var = s2 * (1.0f / DIM) - mu * mu;
        mu_s[tid] = mu;
        rs_s[tid] = rsqrtf(var + 1e-5f);
    }
    __syncthreads();

    // Affine normalize + store (overwrites this block's agg rows in d_out).
    for (int idx = tid; idx < 16 * DIM; idx += 128) {
        int node = idx >> 7;
        int d    = idx & (DIM - 1);
        int n    = nodeBase + node;
        if (n < Ntot)
            out[(size_t)n * DIM + d] =
                (xs[node][d] - mu_s[node]) * rs_s[node] * gamma[d] + beta[d];
    }
}

// ---------------------------------------------------------------------------
extern "C" void kernel_launch(void* const* d_in, const int* in_sizes, int n_in,
                              void* d_out, int out_size, void* d_ws, size_t ws_size,
                              hipStream_t stream) {
    const float* emb   = (const float*)d_in[0];
    const int*   edges = (const int*)  d_in[1];   // [2, E] flattened
    const float* W     = (const float*)d_in[2];
    const float* b     = (const float*)d_in[3];
    const float* gamma = (const float*)d_in[4];
    const float* beta  = (const float*)d_in[5];

    const int E = in_sizes[1] / 2;
    const int N = in_sizes[0] / DIM;
    const int nW = in_sizes[2];                   // 128*256

    const int* row = edges;        // destination node per edge
    const int* col = edges + E;    // source node per edge

    float* out    = (float*)d_out;   // agg accumulator, then final x (in place)
    float* denomW = (float*)d_ws;    // N floats
    // f16 copy of W right after denom, 256B-aligned
    size_t whOff = ((size_t)N * sizeof(float) + 255) & ~(size_t)255;
    _Float16* Wh = (_Float16*)((char*)d_ws + whOff);

    gat_prep_kernel<<<2048, 256, 0, stream>>>(out, denomW, W, Wh, N * DIM, N, nW);

    int edgeBlocks = (E + 7) / 8;    // 8 waves of 32 per block, 1 wave/edge
    gat_edge_kernel<<<edgeBlocks, 256, 0, stream>>>(emb, row, col, denomW, out, E);

    int nodeBlocks = (N + 15) / 16;
    gat_node_kernel<<<nodeBlocks, 128, 0, stream>>>(emb, Wh, b, gamma, beta,
                                                    denomW, out, N);
}
